// MLDecoder_7773890805941
// MI455X (gfx1250) — compile-verified, matrix operand order checked by reading, not run on hardware
//
#include <hip/hip_runtime.h>
#include <math.h>

// ---------------------------------------------------------------------------
// CDNA5 feature detection (compile-safe: falls back if builtins are absent)
// ---------------------------------------------------------------------------
#if defined(__AMDGCN__) && __has_builtin(__builtin_amdgcn_global_load_async_to_lds_b128) && \
    __has_builtin(__builtin_amdgcn_s_wait_asynccnt)
#define HAVE_ASYNC 1
#else
#define HAVE_ASYNC 0
#endif

#if HAVE_ASYNC
typedef int v4i_async __attribute__((vector_size(16)));
typedef __attribute__((address_space(1))) v4i_async* GlobalV4;
typedef __attribute__((address_space(3))) v4i_async* LdsV4;
#endif

// ---------------------------------------------------------------------------
// Types for CDNA5 WMMA (wave32)
// ---------------------------------------------------------------------------
typedef __attribute__((ext_vector_type(16))) __bf16 v16bf;
typedef __attribute__((ext_vector_type(8)))  float  v8f;

union ABFrag { v16bf v; uint4 q[2]; };
union U16x16 { uint4 q[2]; unsigned short s[16]; };

__device__ __forceinline__ unsigned short f2bf(float f) {
  unsigned u = __float_as_uint(f);
  u += 0x7FFFu + ((u >> 16) & 1u);          // round-to-nearest-even
  return (unsigned short)(u >> 16);
}

// Branchless 16-element load (fp32 or bf16) -> 16 bf16 bits; elems >= valid
// are zero-masked with VALU selects. Caller guarantees the 16-elem window
// stays inside the allocation (rows clamped; buffers padded).
template <typename T>
__device__ __forceinline__ void load16(const T* src, int valid, U16x16& u);

template <>
__device__ __forceinline__ void load16<unsigned short>(const unsigned short* src,
                                                       int valid, U16x16& u) {
  u.q[0] = *(const uint4*)(src);
  u.q[1] = *(const uint4*)(src + 8);
  if (valid < 16) {
#pragma unroll
    for (int j = 0; j < 16; ++j)
      if (j >= valid) u.s[j] = 0;
  }
}

template <>
__device__ __forceinline__ void load16<float>(const float* src, int valid, U16x16& u) {
  const float4* p = (const float4*)src;
  float4 a = p[0], b = p[1], c = p[2], d = p[3];
  u.s[0]  = f2bf(a.x); u.s[1]  = f2bf(a.y); u.s[2]  = f2bf(a.z); u.s[3]  = f2bf(a.w);
  u.s[4]  = f2bf(b.x); u.s[5]  = f2bf(b.y); u.s[6]  = f2bf(b.z); u.s[7]  = f2bf(b.w);
  u.s[8]  = f2bf(c.x); u.s[9]  = f2bf(c.y); u.s[10] = f2bf(c.z); u.s[11] = f2bf(c.w);
  u.s[12] = f2bf(d.x); u.s[13] = f2bf(d.y); u.s[14] = f2bf(d.z); u.s[15] = f2bf(d.w);
  if (valid < 16) {
#pragma unroll
    for (int j = 0; j < 16; ++j)
      if (j >= valid) u.s[j] = 0;
  }
}

template <typename T>
__device__ __forceinline__ void stage16(const T* src, int valid, unsigned short* lds) {
  U16x16 u;
  load16<T>(src, valid, u);
  *(uint4*)(lds)     = u.q[0];
  *(uint4*)(lds + 8) = u.q[1];
}

// Stage one 16-bf16 segment: async copy global->LDS when possible (bf16 source,
// full segment), else masked register path.
template <typename T>
__device__ __forceinline__ void stageSeg(const T* src, int valid, unsigned short* lds) {
#if HAVE_ASYNC
  if constexpr (sizeof(T) == 2) {
    if (valid >= 16) {
      __builtin_amdgcn_global_load_async_to_lds_b128((GlobalV4)(const void*)src,
                                                     (LdsV4)(void*)lds, 0, 0);
      __builtin_amdgcn_global_load_async_to_lds_b128((GlobalV4)(const void*)src,
                                                     (LdsV4)(void*)lds, 16, 0);
      return;
    }
  }
#endif
  stage16<T>(src, valid, lds);
}

__device__ __forceinline__ void asyncFence() {
#if HAVE_ASYNC
  __builtin_amdgcn_s_wait_asynccnt(0);
#endif
}

__device__ __forceinline__ void storeC(float* p, float v)          { *p = v; }
__device__ __forceinline__ void storeC(unsigned short* p, float v) { *p = f2bf(v); }

// ---------------------------------------------------------------------------
// Generic batched WMMA GEMM: C = epi(alpha * A@B + bias [+ resid])
//   block tile 128x128, 8 waves (4 along M x 2 along N), wave tile 32x64
//   K-step 32, double-buffered bf16 LDS tiles, f32 accum via
//   v_wmma_f32_16x16x32_bf16. BT=true: B stored [N][K] row-major.
//   batch z decomposed as (b, h) = (z/heads, z%heads)
// ---------------------------------------------------------------------------
#define EPI_NONE  0
#define EPI_RELU  2
#define EPI_GELU  3
#define EPI_RESID 4

template <typename TA, typename TB, typename TC, int EPI, bool BT>
__global__ __launch_bounds__(256) void gemm_wmma(
    const TA* __restrict__ A, const TB* __restrict__ B, TC* C,
    const float* __restrict__ bias, const float* resid,
    int M, int N, int K, int lda, int ldb, int ldc, int heads,
    long long sAb, long long sAh, long long sBb, long long sBh,
    long long sCb, long long sCh, float alpha) {
  __shared__ unsigned short As[2][5120];   // 128 rows x 40 (32 + pad)
  __shared__ unsigned short Bs[2][5120];

  const int tid = threadIdx.x;
  const int z = blockIdx.z;
  const int bb = z / heads, hh = z - bb * heads;
  A += (long long)bb * sAb + (long long)hh * sAh;
  B += (long long)bb * sBb + (long long)hh * sBh;
  const long long cOff = (long long)bb * sCb + (long long)hh * sCh;

  const int m0 = blockIdx.y * 128, n0 = blockIdx.x * 128;
  const int lane = tid & 31, wid = tid >> 5;
  const int wm = (wid & 3) * 32, wn = (wid >> 2) * 64;
  const int l15 = lane & 15, half8 = (lane >> 4) << 3;

  // Per-thread staging coordinates (rows clamped so loads never leave buffer)
  const int ar  = tid >> 1, ac0 = (tid & 1) << 4;           // A: row, k-seg
  const int agm = m0 + ar;
  const long long aRow = (long long)((agm < M) ? agm : 0) * lda;
  const int br  = tid >> 1, bc0 = (tid & 1) << 4;           // B (BT): row, k-seg
  const int bgn = n0 + br;
  const long long bRowT = (long long)((bgn < N) ? bgn : 0) * ldb;
  const int bkk = tid >> 3, bn0 = (tid & 7) << 4;           // B (KN): k-row, n-seg

  auto stageA = [&](int k0, int pbuf) {
    int validA = (agm < M) ? (K - (k0 + ac0)) : 0;
    stageSeg<TA>(A + aRow + (k0 + ac0), validA, &As[pbuf][ar * 40 + ac0]);
  };
  auto stageB = [&](int k0, int pbuf) {
    if constexpr (BT) {  // B is [N][K]: direct row copy into Bs[n][k]
      int validB = (bgn < N) ? (K - (k0 + bc0)) : 0;
      stageSeg<TB>(B + bRowT + (k0 + bc0), validB, &Bs[pbuf][br * 40 + bc0]);
    } else {  // B is [K][N]: load contiguous along n, scatter-transpose
      int gk = k0 + bkk;
      long long bRow = (long long)((gk < K) ? gk : 0) * ldb;
      int validB = (gk < K) ? (N - (n0 + bn0)) : 0;
      U16x16 u;
      load16<TB>(B + bRow + (n0 + bn0), validB, u);
#pragma unroll
      for (int j = 0; j < 16; ++j) Bs[pbuf][(bn0 + j) * 40 + bkk] = u.s[j];
      if (gk + 32 < K)
        __builtin_prefetch(B + (long long)(gk + 32) * ldb + (n0 + bn0), 0, 3);
    }
  };

  v8f acc[2][4];
#pragma unroll
  for (int i = 0; i < 2; ++i)
#pragma unroll
    for (int j = 0; j < 4; ++j)
      acc[i][j] = (v8f){0.f, 0.f, 0.f, 0.f, 0.f, 0.f, 0.f, 0.f};

  // prologue: stage tile 0
  stageA(0, 0);
  stageB(0, 0);
  asyncFence();
  __syncthreads();

  int cur = 0;
  for (int k0 = 0; k0 < K; k0 += 32) {
    // stage next tile into the other buffer while computing this one
    if (k0 + 32 < K) {
      stageA(k0 + 32, cur ^ 1);
      stageB(k0 + 32, cur ^ 1);
    }

    // fragment loads per ISA 7.12.2: lanes 0-15 hold K {0..7,16..23},
    // lanes 16-31 hold K {8..15,24..31}; rows/cols = lane&15
    ABFrag af[2], bf[4];
#pragma unroll
    for (int i = 0; i < 2; ++i) {
      const unsigned short* p = &As[cur][(wm + i * 16 + l15) * 40 + half8];
      af[i].q[0] = *(const uint4*)p;
      af[i].q[1] = *(const uint4*)(p + 16);
    }
#pragma unroll
    for (int j = 0; j < 4; ++j) {
      const unsigned short* p = &Bs[cur][(wn + j * 16 + l15) * 40 + half8];
      bf[j].q[0] = *(const uint4*)p;
      bf[j].q[1] = *(const uint4*)(p + 16);
    }
#pragma unroll
    for (int i = 0; i < 2; ++i)
#pragma unroll
      for (int j = 0; j < 4; ++j)
        acc[i][j] = __builtin_amdgcn_wmma_f32_16x16x32_bf16(
            false, af[i].v, false, bf[j].v, (short)0, acc[i][j], false, false);

    asyncFence();
    __syncthreads();
    cur ^= 1;
  }

  // epilogue: lane holds col n = l15, rows m = vgpr + (lane>=16 ? 8 : 0)
  TC* Cb = C + cOff;
  const float* Rb = resid ? resid + cOff : nullptr;
#pragma unroll
  for (int j = 0; j < 4; ++j) {
    int gn = n0 + wn + j * 16 + l15;
    if (gn >= N) continue;
    float bv = 0.f;
    if constexpr (EPI != EPI_NONE) bv = bias ? bias[gn] : 0.f;
#pragma unroll
    for (int i = 0; i < 2; ++i) {
#pragma unroll
      for (int vi = 0; vi < 8; ++vi) {
        int gm = m0 + wm + i * 16 + half8 + vi;
        if (gm >= M) continue;
        float v = acc[i][j][vi] * alpha + bv;
        if constexpr (EPI == EPI_RELU) v = fmaxf(v, 0.f);
        if constexpr (EPI == EPI_GELU) {
          float t = 0.7978845608028654f * (v + 0.044715f * v * v * v);
          v = 0.5f * v * (1.f + tanhf(t));
        }
        if constexpr (EPI == EPI_RESID) v += Rb[(long long)gm * ldc + gn];
        storeC(Cb + (long long)gm * ldc + gn, v);
      }
    }
  }
}

// ---------------------------------------------------------------------------
// Small kernels
// ---------------------------------------------------------------------------
// w [K][N] fp32 -> wT [N][K] bf16 (tiled transpose; K,N multiples of 32)
__global__ __launch_bounds__(256) void convert_transpose_w(const float* __restrict__ w,
                                                           unsigned short* __restrict__ wT,
                                                           int K, int N) {
  __shared__ float tile[32][33];
  int k0 = blockIdx.x * 32, n0 = blockIdx.y * 32;
  int tx = threadIdx.x, ty = threadIdx.y;  // (32,8)
#pragma unroll
  for (int kk = 0; kk < 32; kk += 8) {
    int k = k0 + ty + kk, n = n0 + tx;
    tile[ty + kk][tx] = (k < K && n < N) ? w[(long long)k * N + n] : 0.f;
  }
  __syncthreads();
#pragma unroll
  for (int kk = 0; kk < 32; kk += 8) {
    int n = n0 + ty + kk, k = k0 + tx;
    if (n < N && k < K) wT[(long long)n * K + k] = f2bf(tile[tx][ty + kk]);
  }
}

// x [64,2048,196] -> xT bf16 [64*196, 2048]
__global__ __launch_bounds__(256) void transpose_x_bf16(const float* __restrict__ x,
                                                        unsigned short* __restrict__ xT) {
  __shared__ float tile[32][33];
  int b = blockIdx.z;
  int c0 = blockIdx.x * 32;  // channel
  int s0 = blockIdx.y * 32;  // spatial
  int tx = threadIdx.x, ty = threadIdx.y;  // (32,8)
#pragma unroll
  for (int kk = 0; kk < 32; kk += 8) {
    int c = c0 + ty + kk, s = s0 + tx;
    tile[ty + kk][tx] = (s < 196) ? x[((long long)b * 2048 + c) * 196 + s] : 0.f;
  }
  __syncthreads();
#pragma unroll
  for (int kk = 0; kk < 32; kk += 8) {
    int s = s0 + ty + kk, c = c0 + tx;
    if (s < 196) xT[((long long)b * 196 + s) * 2048 + c] = f2bf(tile[tx][ty + kk]);
  }
}

// h[(b*100+g)*768+d] = query_embed[g*768+d]
__global__ __launch_bounds__(256) void init_h(const float* __restrict__ qe,
                                              float* __restrict__ h, int total) {
  int i = blockIdx.x * 256 + threadIdx.x;
  if (i < total) h[i] = qe[i % (100 * 768)];
}

// row-wise LayerNorm over 768 (one 256-thread block per row)
__global__ __launch_bounds__(256) void layernorm768(const float* __restrict__ x,
                                                    const float* __restrict__ g,
                                                    const float* __restrict__ b,
                                                    float* __restrict__ y) {
  __shared__ float red[256];
  const int row = blockIdx.x, t = threadIdx.x;
  const float* xr = x + (long long)row * 768;
  float v0 = xr[t], v1 = xr[t + 256], v2 = xr[t + 512];
  red[t] = v0 + v1 + v2;
  __syncthreads();
  for (int o = 128; o > 0; o >>= 1) { if (t < o) red[t] += red[t + o]; __syncthreads(); }
  float mean = red[0] * (1.f / 768.f);
  __syncthreads();
  float d0 = v0 - mean, d1 = v1 - mean, d2 = v2 - mean;
  red[t] = d0 * d0 + d1 * d1 + d2 * d2;
  __syncthreads();
  for (int o = 128; o > 0; o >>= 1) { if (t < o) red[t] += red[t + o]; __syncthreads(); }
  float rs = rsqrtf(red[0] * (1.f / 768.f) + 1e-5f);
  float* yr = y + (long long)row * 768;
  yr[t]       = g[t]       * d0 * rs + b[t];
  yr[t + 256] = g[t + 256] * d1 * rs + b[t + 256];
  yr[t + 512] = g[t + 512] * d2 * rs + b[t + 512];
}

// softmax over nk (<=196), fp32 scores (leading dim ldp) -> bf16 probs (padded)
__global__ __launch_bounds__(256) void softmax_rows(const float* __restrict__ s,
                                                    unsigned short* __restrict__ p,
                                                    int nk, int ldp) {
  __shared__ float red[256];
  int row = blockIdx.x, t = threadIdx.x;
  const float* sr = s + (long long)row * ldp;
  float v = (t < nk) ? sr[t] : -3.0e38f;
  red[t] = v;
  __syncthreads();
  for (int o = 128; o > 0; o >>= 1) { if (t < o) red[t] = fmaxf(red[t], red[t + o]); __syncthreads(); }
  float mx = red[0];
  __syncthreads();
  float e = (t < nk) ? __expf(v - mx) : 0.f;
  red[t] = e;
  __syncthreads();
  for (int o = 128; o > 0; o >>= 1) { if (t < o) red[t] += red[t + o]; __syncthreads(); }
  float inv = 1.f / red[0];
  if (t < ldp) p[(long long)row * ldp + t] = (t < nk) ? f2bf(e * inv) : (unsigned short)0;
}

// logits[b, g*10+f] = sum_d h[b,g,d]*pool[g,d,f] + bias ; one wave per (b,g)
__global__ __launch_bounds__(256) void group_fc(const float* __restrict__ h,
                                                const float* __restrict__ pool,
                                                const float* __restrict__ bias,
                                                float* __restrict__ out) {
  int wid = (blockIdx.x * 256 + threadIdx.x) >> 5;  // 0..6399
  int lane = threadIdx.x & 31;
  int b = wid / 100, g = wid - b * 100;
  const float* hr = h + (long long)wid * 768;
  const float* pr = pool + (long long)g * 7680;
  float acc[10];
#pragma unroll
  for (int f = 0; f < 10; ++f) acc[f] = 0.f;
  for (int d = lane; d < 768; d += 32) {
    float hv = hr[d];
    const float* pp = pr + d * 10;
#pragma unroll
    for (int f = 0; f < 10; ++f) acc[f] += hv * pp[f];
  }
#pragma unroll
  for (int f = 0; f < 10; ++f)
    for (int o = 16; o > 0; o >>= 1) acc[f] += __shfl_xor(acc[f], o, 32);
  if (lane == 0) {
    int base = g * 10;
#pragma unroll
    for (int f = 0; f < 10; ++f)
      out[(long long)b * 1000 + base + f] = acc[f] + bias[base + f];
  }
}

// ---------------------------------------------------------------------------
// Launch
// ---------------------------------------------------------------------------
typedef unsigned short us;

extern "C" void kernel_launch(void* const* d_in, const int* in_sizes, int n_in,
                              void* d_out, int out_size, void* d_ws, size_t ws_size,
                              hipStream_t stream) {
  (void)in_sizes; (void)n_in; (void)out_size; (void)ws_size;
  const float* x           = (const float*)d_in[0];
  const float* W_embed     = (const float*)d_in[1];
  const float* b_embed     = (const float*)d_in[2];
  const float* query_embed = (const float*)d_in[3];
  const float* ln1_g = (const float*)d_in[4];  const float* ln1_b = (const float*)d_in[5];
  const float* Wq_s  = (const float*)d_in[6];  const float* Wk_s  = (const float*)d_in[7];
  const float* Wv_s  = (const float*)d_in[8];  const float* Wo_s  = (const float*)d_in[9];
  const float* bo_s  = (const float*)d_in[10];
  const float* ln2_g = (const float*)d_in[11]; const float* ln2_b = (const float*)d_in[12];
  const float* Wq_c  = (const float*)d_in[13]; const float* Wk_c  = (const float*)d_in[14];
  const float* Wv_c  = (const float*)d_in[15]; const float* Wo_c  = (const float*)d_in[16];
  const float* bo_c  = (const float*)d_in[17];
  const float* ln3_g = (const float*)d_in[18]; const float* ln3_b = (const float*)d_in[19];
  const float* W1 = (const float*)d_in[20]; const float* b1 = (const float*)d_in[21];
  const float* W2 = (const float*)d_in[22]; const float* b2 = (const float*)d_in[23];
  const float* dup_pool = (const float*)d_in[24];
  const float* dup_bias = (const float*)d_in[25];
  float* logits = (float*)d_out;

  const int BS = 64, HW = 196, Cch = 2048, D = 768, G = 100, INNER = 6144,
            HEADS = 8, FF = 2048;
  const int Mx = BS * HW;   // 12544
  const int Mh = BS * G;    // 6400
  const int NKP = 208;      // padded score leading dim (16-elem aligned)

  char* ws = (char*)d_ws;
  size_t off = 0;
  auto alloc = [&](size_t bytes) -> char* {
    char* p = ws + off;
    off = (off + bytes + 255) & ~(size_t)255;
    return p;
  };

  us*    xT  = (us*)alloc((size_t)Mx * Cch * 2);
  us*    emb = (us*)alloc((size_t)Mx * D * 2);
  float* h   = (float*)alloc((size_t)Mh * D * 4);
  float* hn  = (float*)alloc((size_t)Mh * D * 4);
  us*    qb  = (us*)alloc((size_t)Mh * INNER * 2);
  us*    kb  = (us*)alloc((size_t)Mx * INNER * 2);
  us*    vb  = (us*)alloc((size_t)Mx * INNER * 2);
  us*    ob  = (us*)alloc((size_t)Mh * INNER * 2);
  float* sc  = (float*)alloc((size_t)BS * HEADS * G * NKP * 4);
  us*    pb  = (us*)alloc((size_t)BS * HEADS * G * NKP * 2);
  us*    ffb = (us*)alloc((size_t)Mh * FF * 2);
  // transposed bf16 weights: wT[n][k]
  us* wEmbT = (us*)alloc((size_t)Cch * D * 2);
  us* wQsT  = (us*)alloc((size_t)D * INNER * 2);
  us* wKsT  = (us*)alloc((size_t)D * INNER * 2);
  us* wVsT  = (us*)alloc((size_t)D * INNER * 2);
  us* wOsT  = (us*)alloc((size_t)INNER * D * 2);
  us* wQcT  = (us*)alloc((size_t)D * INNER * 2);
  us* wKcT  = (us*)alloc((size_t)D * INNER * 2);
  us* wVcT  = (us*)alloc((size_t)D * INNER * 2);
  us* wOcT  = (us*)alloc((size_t)INNER * D * 2);
  us* wW1T  = (us*)alloc((size_t)D * FF * 2);
  us* wW2T  = (us*)alloc((size_t)FF * D * 2);
  (void)alloc(1024);  // tail pad so clamped vector loads never leave d_ws

  dim3 blk(256);
  dim3 tb32(32, 8);
  auto cvtT = [&](const float* w, us* wT, int K, int N) {
    convert_transpose_w<<<dim3(K / 32, N / 32, 1), tb32, 0, stream>>>(w, wT, K, N);
  };
  cvtT(W_embed, wEmbT, Cch, D);
  cvtT(Wq_s, wQsT, D, INNER); cvtT(Wk_s, wKsT, D, INNER); cvtT(Wv_s, wVsT, D, INNER);
  cvtT(Wo_s, wOsT, INNER, D);
  cvtT(Wq_c, wQcT, D, INNER); cvtT(Wk_c, wKcT, D, INNER); cvtT(Wv_c, wVcT, D, INNER);
  cvtT(Wo_c, wOcT, INNER, D);
  cvtT(W1, wW1T, D, FF); cvtT(W2, wW2T, FF, D);

  // x -> token-major bf16
  {
    dim3 tg(Cch / 32, (HW + 31) / 32, BS);
    transpose_x_bf16<<<tg, tb32, 0, stream>>>(x, xT);
  }

  const float alpha = 1.0f / sqrtf(768.0f);

  // emb = relu(xT @ W_embed + b_embed)  [12544 x 768]
  gemm_wmma<us, us, us, EPI_RELU, true><<<dim3(D / 128, (Mx + 127) / 128, 1), blk, 0, stream>>>(
      xT, wEmbT, emb, b_embed, nullptr, Mx, D, Cch, Cch, Cch, D, 1, 0, 0, 0, 0, 0, 0, 1.0f);

  // h = broadcast(query_embed)
  init_h<<<(Mh * D + 255) / 256, blk, 0, stream>>>(query_embed, h, Mh * D);

  // ---- self attention ----
  layernorm768<<<Mh, blk, 0, stream>>>(h, ln1_g, ln1_b, hn);
  gemm_wmma<float, us, us, EPI_NONE, true><<<dim3(INNER / 128, (Mh + 127) / 128, 1), blk, 0, stream>>>(
      hn, wQsT, qb, nullptr, nullptr, Mh, INNER, D, D, D, INNER, 1, 0, 0, 0, 0, 0, 0, 1.0f);
  gemm_wmma<float, us, us, EPI_NONE, true><<<dim3(INNER / 128, (Mh + 127) / 128, 1), blk, 0, stream>>>(
      hn, wKsT, kb, nullptr, nullptr, Mh, INNER, D, D, D, INNER, 1, 0, 0, 0, 0, 0, 0, 1.0f);
  gemm_wmma<float, us, us, EPI_NONE, true><<<dim3(INNER / 128, (Mh + 127) / 128, 1), blk, 0, stream>>>(
      hn, wVsT, vb, nullptr, nullptr, Mh, INNER, D, D, D, INNER, 1, 0, 0, 0, 0, 0, 0, 1.0f);
  // scores = alpha * q @ k^T   per (b,head)
  gemm_wmma<us, us, float, EPI_NONE, true><<<dim3(1, 1, BS * HEADS), blk, 0, stream>>>(
      qb, kb, sc, nullptr, nullptr, G, G, 768, INNER, INNER, NKP, HEADS,
      (long long)G * INNER, 768, (long long)G * INNER, 768,
      (long long)HEADS * G * NKP, (long long)G * NKP, alpha);
  softmax_rows<<<BS * HEADS * G, blk, 0, stream>>>(sc, pb, G, NKP);
  // o = probs @ v  (B is [K][N] slice of vb)
  gemm_wmma<us, us, us, EPI_NONE, false><<<dim3(768 / 128, 1, BS * HEADS), blk, 0, stream>>>(
      pb, vb, ob, nullptr, nullptr, G, 768, G, NKP, INNER, INNER, HEADS,
      (long long)HEADS * G * NKP, (long long)G * NKP,
      (long long)G * INNER, 768, (long long)G * INNER, 768, 1.0f);
  // h = h + o @ Wo_s + bo_s
  gemm_wmma<us, us, float, EPI_RESID, true><<<dim3(D / 128, (Mh + 127) / 128, 1), blk, 0, stream>>>(
      ob, wOsT, h, bo_s, h, Mh, D, INNER, INNER, INNER, D, 1, 0, 0, 0, 0, 0, 0, 1.0f);

  // ---- cross attention ----
  layernorm768<<<Mh, blk, 0, stream>>>(h, ln2_g, ln2_b, hn);
  gemm_wmma<float, us, us, EPI_NONE, true><<<dim3(INNER / 128, (Mh + 127) / 128, 1), blk, 0, stream>>>(
      hn, wQcT, qb, nullptr, nullptr, Mh, INNER, D, D, D, INNER, 1, 0, 0, 0, 0, 0, 0, 1.0f);
  gemm_wmma<us, us, us, EPI_NONE, true><<<dim3(INNER / 128, (Mx + 127) / 128, 1), blk, 0, stream>>>(
      emb, wKcT, kb, nullptr, nullptr, Mx, INNER, D, D, D, INNER, 1, 0, 0, 0, 0, 0, 0, 1.0f);
  gemm_wmma<us, us, us, EPI_NONE, true><<<dim3(INNER / 128, (Mx + 127) / 128, 1), blk, 0, stream>>>(
      emb, wVcT, vb, nullptr, nullptr, Mx, INNER, D, D, D, INNER, 1, 0, 0, 0, 0, 0, 0, 1.0f);
  gemm_wmma<us, us, float, EPI_NONE, true><<<dim3((HW + 127) / 128, 1, BS * HEADS), blk, 0, stream>>>(
      qb, kb, sc, nullptr, nullptr, G, HW, 768, INNER, INNER, NKP, HEADS,
      (long long)G * INNER, 768, (long long)HW * INNER, 768,
      (long long)HEADS * G * NKP, (long long)G * NKP, alpha);
  softmax_rows<<<BS * HEADS * G, blk, 0, stream>>>(sc, pb, HW, NKP);
  gemm_wmma<us, us, us, EPI_NONE, false><<<dim3(768 / 128, 1, BS * HEADS), blk, 0, stream>>>(
      pb, vb, ob, nullptr, nullptr, G, 768, HW, NKP, INNER, INNER, HEADS,
      (long long)HEADS * G * NKP, (long long)G * NKP,
      (long long)HW * INNER, 768, (long long)G * INNER, 768, 1.0f);
  gemm_wmma<us, us, float, EPI_RESID, true><<<dim3(D / 128, (Mh + 127) / 128, 1), blk, 0, stream>>>(
      ob, wOcT, h, bo_c, h, Mh, D, INNER, INNER, INNER, D, 1, 0, 0, 0, 0, 0, 0, 1.0f);

  // ---- feed-forward ----
  layernorm768<<<Mh, blk, 0, stream>>>(h, ln3_g, ln3_b, hn);
  gemm_wmma<float, us, us, EPI_GELU, true><<<dim3(FF / 128, (Mh + 127) / 128, 1), blk, 0, stream>>>(
      hn, wW1T, ffb, b1, nullptr, Mh, FF, D, D, D, FF, 1, 0, 0, 0, 0, 0, 0, 1.0f);
  gemm_wmma<us, us, float, EPI_RESID, true><<<dim3(D / 128, (Mh + 127) / 128, 1), blk, 0, stream>>>(
      ffb, wW2T, h, b2, h, Mh, D, FF, FF, FF, D, 1, 0, 0, 0, 0, 0, 0, 1.0f);

  // ---- group FC -> logits ----
  group_fc<<<Mh / 8, blk, 0, stream>>>(h, dup_pool, dup_bias, logits);
}
